// CRFAspectSent_1503238553715
// MI455X (gfx1250) — matrix operand order repaired and checked by reading, not compiled, not run on hardware
//
#include <hip/hip_runtime.h>
#include <hip/hip_bf16.h>
#include <math.h>

typedef __attribute__((ext_vector_type(16))) _Float16 v16h;
typedef __attribute__((ext_vector_type(8)))  _Float16 v8h;
typedef __attribute__((ext_vector_type(8)))  float    v8f;

// Problem dims
static constexpr int BB   = 64;
static constexpr int LLEN = 256;
static constexpr int EE   = 300;
static constexpr int MM   = 50;
static constexpr int HH   = 256;   // context dim
static constexpr int HD   = 128;   // per-direction hidden
static constexpr int GD   = 512;   // 4*HD gates
static constexpr int DRAW = 350;   // E+M
static constexpr int DPAD = 352;   // padded to multiple of 32 for WMMA K
static constexpr int ROWS = BB * LLEN; // 16384

// ---------------- WMMA fragment loaders (CDNA5 16x16x32 f16 layouts) -------

// A: 16x32 f16, row-major source with leading dim lda.
// lane<16: row=mb+lane, k = kb+{0..7,16..23}; lane>=16: row=mb+lane-16, k=kb+{8..15,24..31}
__device__ __forceinline__ v16h load_a_frag(const _Float16* A, int lda, int mb, int kb) {
  int lane = threadIdx.x & 31;
  const _Float16* p = A + (size_t)(mb + (lane & 15)) * lda + kb + ((lane >> 4) ? 8 : 0);
  v8h lo = *(const v8h*)p;          // k+0..7   (or +8..15)
  v8h hi = *(const v8h*)(p + 16);   // k+16..23 (or +24..31)
  v16h a;
  #pragma unroll
  for (int j = 0; j < 8; ++j) { a[j] = lo[j]; a[j + 8] = hi[j]; }
  return a;
}

// B: 32x16 f16 where B[k][n] = W[n][k], W row-major (N x K) with leading dim ldw.
// lane<16: col=nb+lane, k=kb+{0..7,16..23}; lane>=16: col=nb+lane-16, k=kb+{8..15,24..31}
__device__ __forceinline__ v16h load_b_fragT(const _Float16* W, int ldw, int nb, int kb) {
  int lane = threadIdx.x & 31;
  const _Float16* p = W + (size_t)(nb + (lane & 15)) * ldw + kb + ((lane >> 4) ? 8 : 0);
  v8h lo = *(const v8h*)p;
  v8h hi = *(const v8h*)(p + 16);
  v16h b;
  #pragma unroll
  for (int j = 0; j < 8; ++j) { b[j] = lo[j]; b[j + 8] = hi[j]; }
  return b;
}

__device__ __forceinline__ v8f zero8() {
  v8f z;
  #pragma unroll
  for (int i = 0; i < 8; ++i) z[i] = 0.0f;
  return z;
}

__device__ __forceinline__ float sigm(float x) { return 1.0f / (1.0f + __expf(-x)); }
__device__ __forceinline__ float lse2(float a, float b) {
  float m = fmaxf(a, b);
  return m + logf(expf(a - m) + expf(b - m));
}

// ---------------- Stage 1: gather embeddings -> f16 x [ROWS][DPAD] ---------
__global__ void pack_x_kernel(const int* __restrict__ sents, const int* __restrict__ masks,
                              const float* __restrict__ wemb, const float* __restrict__ memb,
                              _Float16* __restrict__ xf) {
  size_t i = (size_t)blockIdx.x * blockDim.x + threadIdx.x;
  size_t total = (size_t)ROWS * DPAD;
  if (i >= total) return;
  int col = (int)(i % DPAD);
  size_t row = i / DPAD;
  float v = 0.0f;
  if (col < EE)        v = wemb[(size_t)sents[row] * EE + col];
  else if (col < DRAW) v = memb[(size_t)masks[row] * MM + (col - EE)];
  xf[i] = (_Float16)v;
}

// ---------------- Stage 2: pack w_ih (both dirs) -> f16 [2][GD][DPAD] ------
__global__ void pack_w_kernel(const float* __restrict__ wf, const float* __restrict__ wb,
                              _Float16* __restrict__ wpad) {
  int i = blockIdx.x * blockDim.x + threadIdx.x;
  int total = 2 * GD * DPAD;
  if (i >= total) return;
  int k = i % DPAD;
  int n = (i / DPAD) % GD;
  int d = i / (DPAD * GD);
  const float* w = d ? wb : wf;
  wpad[i] = (_Float16)((k < DRAW) ? w[(size_t)n * DRAW + k] : 0.0f);
}

// ---------------- Stage 3: input GEMM xs = x @ w_ihT + b_ih (WMMA) ---------
// One wave computes a 16(M) x 64(N) tile (A-fragment reuse). 2*1024*8 waves.
__global__ __launch_bounds__(256) void ingemm_kernel(const _Float16* __restrict__ xf,
                                                     const _Float16* __restrict__ wih,
                                                     const float* __restrict__ bih_f,
                                                     const float* __restrict__ bih_b,
                                                     _Float16* __restrict__ xs) {
  int gwave = blockIdx.x * 8 + (threadIdx.x >> 5);
  int nt4 = gwave & 7;
  int mt  = (gwave >> 3) & 1023;
  int dir = gwave >> 13;
  const _Float16* W = wih + (size_t)dir * GD * DPAD;
  const float* bih = dir ? bih_b : bih_f;
  int mbase = mt * 16;
  int nbase0 = nt4 * 64;
  int lane = threadIdx.x & 31;

  v8f acc[4];
  #pragma unroll
  for (int q = 0; q < 4; ++q) acc[q] = zero8();

  for (int kb = 0; kb < DPAD; kb += 32) {
    v16h a = load_a_frag(xf, DPAD, mbase, kb);
    #pragma unroll
    for (int q = 0; q < 4; ++q) {
      v16h b = load_b_fragT(W, DPAD, nbase0 + q * 16, kb);
      acc[q] = __builtin_amdgcn_wmma_f32_16x16x32_f16(false, a, false, b,
                                                      (short)0, acc[q], false, false);
    }
  }

  int rowoff = (lane >> 4) ? 8 : 0;
  _Float16* out = xs + (size_t)dir * ROWS * GD;
  #pragma unroll
  for (int q = 0; q < 4; ++q) {
    int ncol = nbase0 + q * 16 + (lane & 15);
    float bv = bih[ncol];
    #pragma unroll
    for (int r = 0; r < 8; ++r) {
      size_t row = (size_t)(mbase + rowoff + r);
      out[row * GD + ncol] = (_Float16)(acc[q][r] + bv);
    }
  }
}

// ---------------- Stage 4: persistent LSTM recurrence (WMMA per step) ------
// grid.x = 2 (direction). 1024 threads = 32 waves: wave = mt(0..3) x nj(0..7).
// Each wave computes the 4 gate tiles for its (16 batch x 16 hidden) block,
// so activations are lane-local. W_hh staged in LDS as f16, h/c live in LDS.
__global__ __launch_bounds__(1024) void lstm_kernel(const float* __restrict__ w_hh_f,
                                                    const float* __restrict__ w_hh_b,
                                                    const float* __restrict__ b_hh_f,
                                                    const float* __restrict__ b_hh_b,
                                                    const _Float16* __restrict__ xs,
                                                    const int* __restrict__ lens,
                                                    float* __restrict__ context) {
  extern __shared__ char smem[];
  _Float16* wl = (_Float16*)smem;                               // [GD][HD] 128 KB
  _Float16* hl = (_Float16*)(smem + (size_t)GD * HD * 2);       // [BB][HD]  16 KB
  float*    cl = (float*)(smem + (size_t)GD * HD * 2 + (size_t)BB * HD * 2); // 32 KB

  int dir = blockIdx.x;
  const float* whh = dir ? w_hh_b : w_hh_f;
  const float* bhh = dir ? b_hh_b : b_hh_f;
  const _Float16* xsd = xs + (size_t)dir * ROWS * GD;
  int tid = threadIdx.x;

  for (int i = tid; i < GD * HD; i += 1024) wl[i] = (_Float16)whh[i];
  for (int i = tid; i < BB * HD; i += 1024) { hl[i] = (_Float16)0.0f; cl[i] = 0.0f; }
  __syncthreads();

  int wave = tid >> 5, lane = tid & 31;
  int mt = wave >> 3;      // batch tile 0..3
  int nj = wave & 7;       // hidden tile 0..7
  int mbase = mt * 16;
  int nbase = nj * 16;
  int col = nbase + (lane & 15);
  int rowoff = (lane >> 4) ? 8 : 0;

  float bh[4];
  #pragma unroll
  for (int g = 0; g < 4; ++g) bh[g] = bhh[g * HD + col];
  int mylen[8];
  #pragma unroll
  for (int r = 0; r < 8; ++r) mylen[r] = lens[mbase + rowoff + r];

  for (int s = 0; s < LLEN; ++s) {
    int t = dir ? (LLEN - 1 - s) : s;

    v8f acc[4];
    #pragma unroll
    for (int g = 0; g < 4; ++g) acc[g] = zero8();

    #pragma unroll
    for (int kb = 0; kb < HD; kb += 32) {
      v16h a = load_a_frag(hl, HD, mbase, kb);
      #pragma unroll
      for (int g = 0; g < 4; ++g) {
        v16h b = load_b_fragT(wl, HD, g * HD + nbase, kb);
        acc[g] = __builtin_amdgcn_wmma_f32_16x16x32_f16(false, a, false, b,
                                                        (short)0, acc[g], false, false);
      }
    }
    __syncthreads();   // all reads of hl done before update

    #pragma unroll
    for (int r = 0; r < 8; ++r) {
      int b = mbase + rowoff + r;
      const _Float16* xrow = xsd + ((size_t)b * LLEN + t) * GD;
      float gi = acc[0][r] + (float)xrow[0 * HD + col] + bh[0];
      float gf = acc[1][r] + (float)xrow[1 * HD + col] + bh[1];
      float gg = acc[2][r] + (float)xrow[2 * HD + col] + bh[2];
      float go = acc[3][r] + (float)xrow[3 * HD + col] + bh[3];
      float cold = cl[b * HD + col];
      float cn = sigm(gf) * cold + sigm(gi) * tanhf(gg);
      float hn = sigm(go) * tanhf(cn);
      bool valid = (t < mylen[r]);
      if (valid) {
        cl[b * HD + col] = cn;
        hl[b * HD + col] = (_Float16)hn;
      }
      context[((size_t)b * LLEN + t) * HH + dir * HD + col] = valid ? hn : 0.0f;
    }
    __syncthreads();   // h/c updated before next step's reads
  }
}

// ---------------- Stage 5: target-average pooling over mask ----------------
__global__ void tavg_kernel(const int* __restrict__ masks, const float* __restrict__ context,
                            float* __restrict__ tavg) {
  int b = blockIdx.x, h = threadIdx.x;
  float acc = 0.0f, ms = 0.0f;
  for (int t = 0; t < LLEN; ++t) {
    float m = (float)masks[b * LLEN + t];
    acc += m * context[((size_t)b * LLEN + t) * HH + h];
    ms += m;
  }
  tavg[b * HH + h] = acc / ms;
}

__global__ void addtavg_kernel(float* __restrict__ context, const float* __restrict__ tavg) {
  size_t i = (size_t)blockIdx.x * blockDim.x + threadIdx.x;
  size_t total = (size_t)ROWS * HH;
  if (i >= total) return;
  int h = (int)(i & (HH - 1));
  int b = (int)(i / ((size_t)HH * LLEN));
  context[i] += tavg[b * HH + h];
}

// ---------------- Stage 6: emissions (B,L,2) --------------------------------
__global__ void emit_kernel(const float* __restrict__ context, const float* __restrict__ w,
                            const float* __restrict__ bias, float* __restrict__ emit) {
  int i = blockIdx.x * blockDim.x + threadIdx.x;
  if (i >= ROWS * 2) return;
  int s = i & 1;
  int rt = i >> 1;
  const float* c = context + (size_t)rt * HH;
  const float* ws = w + s * HH;
  float acc = bias[s];
  for (int k = 0; k < HH; ++k) acc += c[k] * ws[k];
  emit[i] = acc;
}

// ---------------- Stage 7: 2-state CRF forward/backward + marginals --------
__global__ void crf_kernel(const float* __restrict__ emit, const int* __restrict__ lens,
                           const float* __restrict__ trans,
                           float* __restrict__ alphas, float* __restrict__ betas,
                           float* __restrict__ sp, float* __restrict__ spsum) {
  int b = threadIdx.x;
  if (b >= BB) return;
  float T00 = trans[0], T01 = trans[1], T10 = trans[2], T11 = trans[3];
  int len = lens[b];

  float a0 = emit[(b * LLEN + 0) * 2 + 0];
  float a1 = emit[(b * LLEN + 0) * 2 + 1];
  alphas[((size_t)0 * BB + b) * 2 + 0] = a0;
  alphas[((size_t)0 * BB + b) * 2 + 1] = a1;
  for (int t = 1; t < LLEN; ++t) {
    if (t < len) {
      float n0 = emit[(b * LLEN + t) * 2 + 0] + lse2(a0 + T00, a1 + T10);
      float n1 = emit[(b * LLEN + t) * 2 + 1] + lse2(a0 + T01, a1 + T11);
      a0 = n0; a1 = n1;
    }
    alphas[((size_t)t * BB + b) * 2 + 0] = a0;
    alphas[((size_t)t * BB + b) * 2 + 1] = a1;
  }
  float logZ = lse2(a0, a1);

  float be0 = 0.0f, be1 = 0.0f;
  betas[((size_t)(LLEN - 1) * BB + b) * 2 + 0] = 0.0f;
  betas[((size_t)(LLEN - 1) * BB + b) * 2 + 1] = 0.0f;
  for (int t = LLEN - 2; t >= 0; --t) {
    if ((t + 1) < len) {
      float e0 = emit[(b * LLEN + t + 1) * 2 + 0] + be0;
      float e1 = emit[(b * LLEN + t + 1) * 2 + 1] + be1;
      float n0 = lse2(T00 + e0, T01 + e1);
      float n1 = lse2(T10 + e0, T11 + e1);
      be0 = n0; be1 = n1;
    }
    betas[((size_t)t * BB + b) * 2 + 0] = be0;
    betas[((size_t)t * BB + b) * 2 + 1] = be1;
  }

  float ss = 0.0f;
  for (int t = 0; t < LLEN; ++t) {
    float v = 0.0f;
    if (t < len) {
      v = expf(alphas[((size_t)t * BB + b) * 2 + 1] +
               betas[((size_t)t * BB + b) * 2 + 1] - logZ);
    }
    sp[b * LLEN + t] = v;
    ss += v;
  }
  spsum[b] = ss;
}

// ---------------- Stage 8: sentence vector ---------------------------------
__global__ void sentv_kernel(const float* __restrict__ sp, const float* __restrict__ context,
                             float* __restrict__ sentv) {
  int b = blockIdx.x, h = threadIdx.x;
  float acc = 0.0f;
  for (int t = 0; t < LLEN; ++t)
    acc += sp[b * LLEN + t] * context[((size_t)b * LLEN + t) * HH + h];
  sentv[b * HH + h] = acc;
}

// ---------------- Stage 9: loss + penalty -----------------------------------
__global__ void final_kernel(const float* __restrict__ sentv, const float* __restrict__ f2l_w,
                             const float* __restrict__ f2l_b, const int* __restrict__ labels,
                             const float* __restrict__ spsum, const float* __restrict__ trans,
                             float* __restrict__ out) {
  __shared__ float red[BB], red2[BB];
  int b = threadIdx.x;
  if (b < BB) {
    float sc[3];
    const float* v = sentv + (size_t)b * HH;
    #pragma unroll
    for (int s = 0; s < 3; ++s) {
      float a = f2l_b[s];
      const float* w = f2l_w + s * HH;
      for (int k = 0; k < HH; ++k) a += v[k] * w[k];
      sc[s] = a;
    }
    float m = fmaxf(sc[0], fmaxf(sc[1], sc[2]));
    float lse = m + logf(expf(sc[0] - m) + expf(sc[1] - m) + expf(sc[2] - m));
    float lp = sc[labels[b]] - lse;
    red[b] = -lp;
    red2[b] = spsum[b];
  }
  __syncthreads();
  if (b == 0) {
    float cl = 0.0f, sn = 0.0f;
    for (int i = 0; i < BB; ++i) { cl += red[i]; sn += red2[i]; }
    cl /= (float)BB; sn /= (float)BB;
    float pena = fmaxf(trans[2] - trans[0], 0.0f) + fmaxf(trans[1] - trans[3], 0.0f);
    out[0] = cl;
    out[1] = 1.0f * pena + 0.1f * sn;
  }
}

// ---------------- Launch ----------------------------------------------------
extern "C" void kernel_launch(void* const* d_in, const int* in_sizes, int n_in,
                              void* d_out, int out_size, void* d_ws, size_t ws_size,
                              hipStream_t stream) {
  (void)in_sizes; (void)n_in; (void)out_size; (void)ws_size;
  const int*   sents   = (const int*)d_in[0];
  const int*   masks   = (const int*)d_in[1];
  const int*   labels  = (const int*)d_in[2];
  const int*   lens    = (const int*)d_in[3];
  const float* wemb    = (const float*)d_in[4];
  const float* memb    = (const float*)d_in[5];
  const float* w_ih_f  = (const float*)d_in[6];
  const float* w_hh_f  = (const float*)d_in[7];
  const float* b_ih_f  = (const float*)d_in[8];
  const float* b_hh_f  = (const float*)d_in[9];
  const float* w_ih_b  = (const float*)d_in[10];
  const float* w_hh_b  = (const float*)d_in[11];
  const float* b_ih_b  = (const float*)d_in[12];
  const float* b_hh_b  = (const float*)d_in[13];
  const float* f2t_w   = (const float*)d_in[14];
  const float* f2t_b   = (const float*)d_in[15];
  const float* trans   = (const float*)d_in[16];
  const float* f2l_w   = (const float*)d_in[17];
  const float* f2l_b   = (const float*)d_in[18];
  float* out = (float*)d_out;

  // Workspace carve-up (256B aligned)
  char* ws = (char*)d_ws;
  size_t off = 0;
  auto carve = [&](size_t bytes) { char* p = ws + off; off += (bytes + 255) & ~(size_t)255; return p; };
  _Float16* xf16    = (_Float16*)carve((size_t)ROWS * DPAD * 2);   // 11.5 MB
  _Float16* wih16   = (_Float16*)carve((size_t)2 * GD * DPAD * 2); // 0.7 MB
  _Float16* xs      = (_Float16*)carve((size_t)2 * ROWS * GD * 2); // 33.5 MB
  float*    context = (float*)carve((size_t)ROWS * HH * 4);        // 16.8 MB
  float*    tavg    = (float*)carve((size_t)BB * HH * 4);
  float*    emit    = (float*)carve((size_t)ROWS * 2 * 4);
  float*    alphas  = (float*)carve((size_t)LLEN * BB * 2 * 4);
  float*    betas   = (float*)carve((size_t)LLEN * BB * 2 * 4);
  float*    sp      = (float*)carve((size_t)BB * LLEN * 4);
  float*    spsum   = (float*)carve((size_t)BB * 4);
  float*    sentv   = (float*)carve((size_t)BB * HH * 4);

  // 1) embeddings -> f16 padded x
  {
    size_t total = (size_t)ROWS * DPAD;
    int blocks = (int)((total + 255) / 256);
    pack_x_kernel<<<blocks, 256, 0, stream>>>(sents, masks, wemb, memb, xf16);
  }
  // 2) pack input weights
  {
    int total = 2 * GD * DPAD;
    pack_w_kernel<<<(total + 255) / 256, 256, 0, stream>>>(w_ih_f, w_ih_b, wih16);
  }
  // 3) WMMA input GEMM, both directions: 16384 waves, 8 waves/block
  ingemm_kernel<<<2048, 256, 0, stream>>>(xf16, wih16, b_ih_f, b_ih_b, xs);
  // 4) persistent WMMA LSTM, one block per direction
  {
    size_t smem = (size_t)GD * HD * 2 + (size_t)BB * HD * 2 + (size_t)BB * HD * 4; // 176 KB
    lstm_kernel<<<2, 1024, smem, stream>>>(w_hh_f, w_hh_b, b_hh_f, b_hh_b, xs, lens, context);
  }
  // 5) target-average + broadcast add
  tavg_kernel<<<BB, HH, 0, stream>>>(masks, context, tavg);
  {
    size_t total = (size_t)ROWS * HH;
    addtavg_kernel<<<(int)((total + 255) / 256), 256, 0, stream>>>(context, tavg);
  }
  // 6) emissions
  emit_kernel<<<(ROWS * 2 + 255) / 256, 256, 0, stream>>>(context, f2t_w, f2t_b, emit);
  // 7) CRF
  crf_kernel<<<1, 64, 0, stream>>>(emit, lens, trans, alphas, betas, sp, spsum);
  // 8) sentence vector
  sentv_kernel<<<BB, HH, 0, stream>>>(sp, context, sentv);
  // 9) loss
  final_kernel<<<1, 64, 0, stream>>>(sentv, f2l_w, f2l_b, labels, spsum, trans, out);
}
